// DeltaNetLayer_62302795596514
// MI455X (gfx1250) — matrix operand order repaired and checked by reading, not compiled, not run on hardware
//
#include <hip/hip_runtime.h>
#include <cstdint>

// ---------------------------------------------------------------------------
// DeltaNet layer for MI455X (gfx1250, wave32, WMMA + async global->LDS DMA).
//   B=4, S=2048, D=1024, NHEAD=16, HEAD=64  ->  M = B*S = 8192 rows.
// GEMMs: v_wmma_f32_16x16x32_bf16, f32 tiles DMA'd to LDS via
// global_load_async_to_lds_b128 (ASYNCcnt), bf16 convert fused into fragment
// construction. Recurrence: M and M^T in VGPRs, q/k async-prefetched to LDS.
// ---------------------------------------------------------------------------

#define D_MODEL 1024
#define NHEAD   16
#define HEAD    64
#define BATCH   4
#define SEQ     2048
#define MROWS   (BATCH * SEQ)          // 8192
#define EPS     1e-6f

typedef __attribute__((ext_vector_type(16))) __bf16 v16bf;
typedef __attribute__((ext_vector_type(8)))  float  v8f;
typedef __attribute__((ext_vector_type(4)))  int    v4i;

__device__ __forceinline__ float silu_f(float v) { return v / (1.f + __expf(-v)); }

// ---- async global->LDS copies (CDNA5 GLOBAL_LOAD_ASYNC_TO_LDS_*) ----------
// Builtin signatures (from compiler diagnostics):
//   b32 : (AS1 int*,  AS3 int*,  imm offset, imm cpol)
//   b128: (AS1 v4i*,  AS3 v4i*,  imm offset, imm cpol)
__device__ __forceinline__ void async_copy_b128(const float* g, float* l) {
    v4i* gv = (v4i*)const_cast<float*>(g);
    v4i* lv = (v4i*)l;
    __builtin_amdgcn_global_load_async_to_lds_b128(
        (__attribute__((address_space(1))) v4i*)gv,
        (__attribute__((address_space(3))) v4i*)lv, 0, 0);
}
__device__ __forceinline__ void async_copy_b32(const float* g, float* l) {
    int* gv = (int*)const_cast<float*>(g);
    int* lv = (int*)l;
    __builtin_amdgcn_global_load_async_to_lds_b32(
        (__attribute__((address_space(1))) int*)gv,
        (__attribute__((address_space(3))) int*)lv, 0, 0);
}
#define WAIT_ASYNC0() asm volatile("s_wait_asynccnt 0x0" ::: "memory")

// Build a 16-elem bf16 fragment from two contiguous 8-float LDS segments.
__device__ __forceinline__ v16bf pack16(const float* a, const float* b) {
    v16bf v;
    #pragma unroll
    for (int i = 0; i < 8; ++i) v[i] = (__bf16)a[i];
    #pragma unroll
    for (int i = 0; i < 8; ++i) v[8 + i] = (__bf16)b[i];
    return v;
}

// ------------------------- WMMA GEMM:  Out = act(X @ W^T) -------------------
// X: [M][K] f32 row-major.  W: [N][K] f32 row-major (B[k][n] = W[n][k]).
// Block tile 128x128, K-step 32, 256 threads = 8 waves; each wave: 64x32 via
// 4x2 fragments. Double-buffered LDS f32 tiles filled by async DMA.
#define LDSF 36   // f32 elems per LDS row (32 + 4 pad) = 144B, 16B multiple

__global__ __launch_bounds__(256) void gemm_wmma_bf16(
    const float* __restrict__ X, const float* __restrict__ W,
    float* __restrict__ Out, int M, int N, int K, int act /*0=none,1=silu*/) {

    __shared__ __align__(16) float Asf[2][128 * LDSF];
    __shared__ __align__(16) float Bsf[2][128 * LDSF];

    const int tid   = threadIdx.x;
    const int lane  = tid & 31;
    const int wave  = tid >> 5;
    const int wm    = wave & 1;          // 0..1 -> 64-row strip
    const int wn    = wave >> 1;         // 0..3 -> 32-col strip
    const int mBase = blockIdx.y * 128;
    const int nBase = blockIdx.x * 128;

    // Per-thread DMA chunks: 4 x 16B per tile (128 rows x 32 f32 = 16KB/tile).
    int crow[4], ccol[4];
    #pragma unroll
    for (int c = 0; c < 4; ++c) {
        int idx = c * 256 + tid;
        crow[c] = idx >> 3;
        ccol[c] = (idx & 7) * 4;
    }

    // Prologue: DMA tile 0 into buffer 0.
    #pragma unroll
    for (int c = 0; c < 4; ++c) {
        async_copy_b128(X + (size_t)(mBase + crow[c]) * K + ccol[c],
                        &Asf[0][crow[c] * LDSF + ccol[c]]);
        async_copy_b128(W + (size_t)(nBase + crow[c]) * K + ccol[c],
                        &Bsf[0][crow[c] * LDSF + ccol[c]]);
    }

    v8f acc[4][2];
    #pragma unroll
    for (int i = 0; i < 4; ++i)
        #pragma unroll
        for (int j = 0; j < 2; ++j)
            acc[i][j] = (v8f){0.f, 0.f, 0.f, 0.f, 0.f, 0.f, 0.f, 0.f};

    const int nK = K >> 5;
    for (int i = 0; i < nK; ++i) {
        const int cur = i & 1;
        WAIT_ASYNC0();         // our tile-i DMA chunks have landed in LDS
        __syncthreads();       // visibility across waves + closes prior reads

        if (i + 1 < nK) {      // overlap tile i+1 DMA with tile i compute
            const int kt = (i + 1) << 5;
            #pragma unroll
            for (int c = 0; c < 4; ++c) {
                async_copy_b128(X + (size_t)(mBase + crow[c]) * K + kt + ccol[c],
                                &Asf[cur ^ 1][crow[c] * LDSF + ccol[c]]);
                async_copy_b128(W + (size_t)(nBase + crow[c]) * K + kt + ccol[c],
                                &Bsf[cur ^ 1][crow[c] * LDSF + ccol[c]]);
            }
        }

        // Fragment construction with fused f32->bf16 convert.
        // A 16x32: lanes 0-15 rows, K {kg..kg+7, 16+kg..16+kg+7}, kg = 8*(lane>=16)
        // B 32x16: lanes 0-15 cols, K {kg..kg+7, 16+kg..16+kg+7}, kg = 8*(lane>=16)
        const int kg = (lane >> 4) << 3;
        const int kh = (lane >> 4) << 4;
        v16bf af[4], bfr[2];
        #pragma unroll
        for (int fm = 0; fm < 4; ++fm) {
            const float* p = &Asf[cur][(wm * 64 + fm * 16 + (lane & 15)) * LDSF + kg];
            af[fm] = pack16(p, p + 16);
        }
        #pragma unroll
        for (int fn = 0; fn < 2; ++fn) {
            const float* p = &Bsf[cur][(wn * 32 + fn * 16 + (lane & 15)) * LDSF + kg];
            bfr[fn] = pack16(p, p + 16);
        }

        #pragma unroll
        for (int fm = 0; fm < 4; ++fm)
            #pragma unroll
            for (int fn = 0; fn < 2; ++fn)
                acc[fm][fn] = __builtin_amdgcn_wmma_f32_16x16x32_bf16(
                    false, af[fm], false, bfr[fn], (short)0, acc[fm][fn], false, false);
    }

    // Store: lanes 0-15 rows r, lanes 16-31 rows 8+r within each 16x16 frag.
    const int rOff = (lane >> 4) << 3;
    #pragma unroll
    for (int fm = 0; fm < 4; ++fm) {
        #pragma unroll
        for (int fn = 0; fn < 2; ++fn) {
            int col  = nBase + wn * 32 + fn * 16 + (lane & 15);
            int rowb = mBase + wm * 64 + fm * 16 + rOff;
            v8f c = acc[fm][fn];
            #pragma unroll
            for (int r = 0; r < 8; ++r) {
                float v = c[r];
                if (act == 1) v = silu_f(v);
                Out[(size_t)(rowb + r) * N + col] = v;
            }
        }
    }
}

// ------------------------- beta = sigmoid(x @ Wbeta^T) ----------------------
__global__ __launch_bounds__(256) void beta_kernel(
    const float* __restrict__ X, const float* __restrict__ Wb,
    float* __restrict__ Beta) {
    int gw   = (blockIdx.x * 256 + threadIdx.x) >> 5;
    int lane = threadIdx.x & 31;
    int m = gw >> 4, n = gw & 15;
    const float* xr = X  + (size_t)m * D_MODEL;
    const float* wr = Wb + (size_t)n * D_MODEL;
    float s = 0.f;
    for (int k = lane; k < D_MODEL; k += 32) s += xr[k] * wr[k];
    #pragma unroll
    for (int off = 16; off > 0; off >>= 1) s += __shfl_xor(s, off, 32);
    if (lane == 0) Beta[(size_t)m * NHEAD + n] = 1.f / (1.f + __expf(-s));
}

// ------------------ per-head L2 normalize (rows of 64), in place ------------
__global__ __launch_bounds__(256) void l2norm_heads(float* __restrict__ T) {
    int w    = (blockIdx.x * 256 + threadIdx.x) >> 5;
    int lane = threadIdx.x & 31;
    float* p = T + (size_t)w * HEAD;
    float a = p[lane], b = p[lane + 32];
    float s = a * a + b * b;
    #pragma unroll
    for (int off = 16; off > 0; off >>= 1) s += __shfl_xor(s, off, 32);
    float inv = 1.f / (sqrtf(s) + EPS);
    p[lane] = a * inv;
    p[lane + 32] = b * inv;
}

// ------------------------- delta-rule recurrence ----------------------------
// One 64-thread block per (b, head). Thread t owns row t of M and of M^T.
// q/k for step s+1 are async-DMA'd into the parity LDS buffer during step s;
// v/beta are software-pipelined in registers. 2 barriers / step.
__global__ __launch_bounds__(64) void delta_recurrence(
    const float* __restrict__ Q, const float* __restrict__ Kn,
    const float* __restrict__ V, const float* __restrict__ Beta,
    float* __restrict__ Out) {

    const int bn  = blockIdx.x;           // b*NHEAD + n
    const int b   = bn >> 4, n = bn & 15;
    const int tid = threadIdx.x;          // 0..63

    __shared__ float sq[2][HEAD], sk[2][HEAD], sd[2][HEAD];

    float Mrow[HEAD], MTrow[HEAD];
    #pragma unroll
    for (int i = 0; i < HEAD; ++i) { Mrow[i] = 0.f; MTrow[i] = 0.f; }

    const size_t base0 = ((size_t)b * SEQ * NHEAD + n) * HEAD;   // s=0, stride 1024/step
    // Prologue: DMA q/k for s=0; preload v/beta for s=0.
    async_copy_b32(Q  + base0 + tid, &sq[0][tid]);
    async_copy_b32(Kn + base0 + tid, &sk[0][tid]);
    float vv = V[base0 + tid];
    float bt = Beta[(size_t)b * SEQ * NHEAD + n];

    for (int s = 0; s < SEQ; ++s) {
        const int p = s & 1;
        const size_t base = base0 + (size_t)s * (NHEAD * HEAD);

        WAIT_ASYNC0();          // q/k of step s are in LDS (our wave's DMAs)
        __syncthreads();        // cross-wave visibility; closes step s-1 reads

        float vnext = 0.f, bnext = 0.f;
        if (s + 1 < SEQ) {      // prefetch step s+1 while we compute step s
            const size_t baseN = base + (NHEAD * HEAD);
            async_copy_b32(Q  + baseN + tid, &sq[p ^ 1][tid]);
            async_copy_b32(Kn + baseN + tid, &sk[p ^ 1][tid]);
            vnext = V[baseN + tid];
            bnext = Beta[((size_t)b * SEQ + s + 1) * NHEAD + n];
        }

        // retrieved[tid] = M[tid,:].k ;  out[tid] = M^T[tid,:].q (pre-update)
        float r0 = 0.f, r1 = 0.f, r2 = 0.f, r3 = 0.f;
        float o0 = 0.f, o1 = 0.f, o2 = 0.f, o3 = 0.f;
        #pragma unroll
        for (int j = 0; j < HEAD; j += 4) {
            r0 += Mrow[j + 0] * sk[p][j + 0];
            r1 += Mrow[j + 1] * sk[p][j + 1];
            r2 += Mrow[j + 2] * sk[p][j + 2];
            r3 += Mrow[j + 3] * sk[p][j + 3];
            o0 += MTrow[j + 0] * sq[p][j + 0];
            o1 += MTrow[j + 1] * sq[p][j + 1];
            o2 += MTrow[j + 2] * sq[p][j + 2];
            o3 += MTrow[j + 3] * sq[p][j + 3];
        }
        float retr = (r0 + r1) + (r2 + r3);
        float o    = (o0 + o1) + (o2 + o3);
        sd[p][tid] = bt * (vv - retr);      // beta * delta
        __syncthreads();

        // rank-1 update, row-local for both M and M^T
        float kh = sk[p][tid], bd = sd[p][tid];
        #pragma unroll
        for (int j = 0; j < HEAD; ++j) {
            Mrow[j]  += kh * sd[p][j];
            MTrow[j] += sk[p][j] * bd;
        }
        Out[base + tid] = o;                // (b,s,n*64+d) == (B,S,D) layout
        vv = vnext; bt = bnext;
    }
}

// ------------------------- RMSNorm over rows of 1024 ------------------------
__global__ __launch_bounds__(256) void rmsnorm_rows(
    const float* __restrict__ In, const float* __restrict__ w,
    float* __restrict__ Outp) {
    const int row = blockIdx.x, tid = threadIdx.x;
    const float* ip = In + (size_t)row * D_MODEL;
    float x0 = ip[tid], x1 = ip[tid + 256], x2 = ip[tid + 512], x3 = ip[tid + 768];
    float s = x0 * x0 + x1 * x1 + x2 * x2 + x3 * x3;
    #pragma unroll
    for (int off = 16; off > 0; off >>= 1) s += __shfl_xor(s, off, 32);
    __shared__ float parts[8];
    __shared__ float totalSh;
    if ((tid & 31) == 0) parts[tid >> 5] = s;
    __syncthreads();
    if (tid == 0) {
        float t = 0.f;
        #pragma unroll
        for (int i = 0; i < 8; ++i) t += parts[i];
        totalSh = t;
    }
    __syncthreads();
    float inv = 1.f / sqrtf(totalSh * (1.f / D_MODEL) + EPS);
    float* op = Outp + (size_t)row * D_MODEL;
    op[tid]       = x0 * inv * w[tid];
    op[tid + 256] = x1 * inv * w[tid + 256];
    op[tid + 512] = x2 * inv * w[tid + 512];
    op[tid + 768] = x3 * inv * w[tid + 768];
}

// ---------------------------------------------------------------------------
extern "C" void kernel_launch(void* const* d_in, const int* in_sizes, int n_in,
                              void* d_out, int out_size, void* d_ws, size_t ws_size,
                              hipStream_t stream) {
    const float* x    = (const float*)d_in[0];
    const float* Wq   = (const float*)d_in[1];
    const float* Wk   = (const float*)d_in[2];
    const float* Wv   = (const float*)d_in[3];
    const float* Wo   = (const float*)d_in[4];
    const float* Wb   = (const float*)d_in[5];
    const float* rmsw = (const float*)d_in[6];
    float* out = (float*)d_out;

    const size_t MD = (size_t)MROWS * D_MODEL;      // 8,388,608 floats
    float* ws     = (float*)d_ws;
    float* Q      = ws;
    float* K      = Q + MD;
    float* V      = K + MD;
    float* Beta   = V + MD;                         // MROWS*NHEAD floats
    float* AttnO  = Beta + (size_t)MROWS * NHEAD;
    float* Normed = AttnO + MD;                     // total ~168 MB

    dim3 gGemm(D_MODEL / 128, MROWS / 128);         // (8, 64)

    gemm_wmma_bf16<<<gGemm, 256, 0, stream>>>(x, Wq, Q, MROWS, D_MODEL, D_MODEL, 1);
    gemm_wmma_bf16<<<gGemm, 256, 0, stream>>>(x, Wk, K, MROWS, D_MODEL, D_MODEL, 1);
    gemm_wmma_bf16<<<gGemm, 256, 0, stream>>>(x, Wv, V, MROWS, D_MODEL, D_MODEL, 1);

    beta_kernel<<<(MROWS * NHEAD) / 8, 256, 0, stream>>>(x, Wb, Beta);

    l2norm_heads<<<(MROWS * NHEAD) / 8, 256, 0, stream>>>(Q);
    l2norm_heads<<<(MROWS * NHEAD) / 8, 256, 0, stream>>>(K);

    delta_recurrence<<<BATCH * NHEAD, HEAD, 0, stream>>>(Q, K, V, Beta, AttnO);

    rmsnorm_rows<<<MROWS, 256, 0, stream>>>(AttnO, rmsw, Normed);
    gemm_wmma_bf16<<<gGemm, 256, 0, stream>>>(Normed, Wo, out, MROWS, D_MODEL, D_MODEL, 0);
}
